// PermNet_23828478558235
// MI455X (gfx1250) — compile-verified, matrix operand order checked by reading, not run on hardware
//
#include <hip/hip_runtime.h>
#include <math.h>

typedef __attribute__((ext_vector_type(16))) __bf16 v16bf;
typedef __attribute__((ext_vector_type(8)))  float  v8f;

constexpr int B_   = 128;
constexpr int N_   = 36;    // boxes/attention length, also LSTM seq length
constexpr int NP_  = 48;    // N padded to 3*16
constexpr int FEAT = 2048;
constexpr int MID  = 517;
constexpr int KP_  = 544;   // MID padded to 17*32
constexpr int HID  = 128;
constexpr int OUT_ = 1024;
constexpr int G4   = 4096;  // 4*OUT

__device__ __forceinline__ float sigf(float x) { return 1.0f / (1.0f + expf(-x)); }

// ---- WMMA helpers -----------------------------------------------------------
// A/B 16x32 bf16 fragment: lane L holds row (L&15); K chunks at 8*(L>>4)+[0,8)
// and +16..+23 relative to tile k0.  Each chunk is 8 bf16 = 16 bytes.
__device__ __forceinline__ v16bf load_ab(const __bf16* tile, int ld, int lane) {
    const __bf16* p = tile + (size_t)(lane & 15) * ld + ((lane >> 4) << 3);
    v16bf r;
#pragma unroll
    for (int i = 0; i < 8; ++i) { r[i] = p[i]; r[i + 8] = p[i + 16]; }
    return r;
}

__device__ __forceinline__ v8f wmma_bf16(v16bf a, v16bf b, v8f c) {
    return __builtin_amdgcn_wmma_f32_16x16x32_bf16(
        /*neg_a=*/false, a, /*neg_b=*/false, b,
        /*c_mod=*/(short)0, c, /*reuse_a=*/false, /*reuse_b=*/false);
}

// ---- conversion / padding ---------------------------------------------------
__global__ void k_padcvt(const float* __restrict__ src, __bf16* __restrict__ dst,
                         int rows, int scols, int dcols) {
    size_t idx = (size_t)blockIdx.x * blockDim.x + threadIdx.x;
    size_t tot = (size_t)rows * dcols;
    if (idx >= tot) return;
    int c = (int)(idx % dcols);
    size_t r = idx / dcols;
    float v = (c < scols) ? src[r * scols + c] : 0.0f;
    dst[idx] = (__bf16)v;
}

// features (B,36,2048) f32 -> (B,48,2048) bf16, rows >= 36 zeroed
__global__ void k_cvt_feat(const float* __restrict__ feat, __bf16* __restrict__ dst) {
    size_t idx = (size_t)blockIdx.x * blockDim.x + threadIdx.x;
    size_t tot = (size_t)B_ * NP_ * FEAT;
    if (idx >= tot) return;
    int k = (int)(idx % FEAT);
    int n = (int)((idx / FEAT) % NP_);
    int b = (int)(idx / ((size_t)FEAT * NP_));
    float v = (n < N_) ? feat[((size_t)b * N_ + n) * FEAT + k] : 0.0f;
    dst[idx] = (__bf16)v;
}

// ---- box normalization + attention rows of feats (transposed layout) --------
// featsT[b][n][m] (bf16, KP_ stride) holds feats[b, m, n]; must be pre-zeroed.
__global__ void k_prep(const float* __restrict__ boxes, const float* __restrict__ att,
                       __bf16* __restrict__ featsT, float* __restrict__ satt) {
    int b = blockIdx.x;
    int t = threadIdx.x;            // 64 threads
    __shared__ float red[64];
    float mx = -1e30f;
    for (int i = t; i < 4 * N_; i += 64) mx = fmaxf(mx, boxes[(size_t)b * 4 * N_ + i]);
    red[t] = mx;
    __syncthreads();
    for (int s = 32; s > 0; s >>= 1) {
        if (t < s) red[t] = fmaxf(red[t], red[t + s]);
        __syncthreads();
    }
    float inv = 1.0f / red[0];
    if (t < N_) {
        int n = t;
        __bf16* row = featsT + ((size_t)b * NP_ + n) * KP_;
        for (int c = 0; c < 4; ++c)
            row[c] = (__bf16)(boxes[((size_t)b * 4 + c) * N_ + n] * inv);
        float a = att[(size_t)b * N_ + n];
        row[4] = (__bf16)a;
        satt[(size_t)b * N_ + n] = sigf(a);
    }
}

// ---- conv GEMM: f[b,m,n] = sum_k conv_w[m,k] * features[b,n,k] + conv_b -----
// one wave per 64x16 tile (4 M-subtiles share each B fragment);
// stores transposed into featsT[b][n][5+m]
__global__ void k_conv_wmma(const __bf16* __restrict__ Wc, const __bf16* __restrict__ Fb,
                            const float* __restrict__ conv_b, __bf16* __restrict__ featsT) {
    int wid  = blockIdx.x * 8 + (threadIdx.x >> 5);   // 128 b * 8 mtg * 3 nt waves
    int lane = threadIdx.x & 31;
    int nt  = wid % 3;
    int mtg = (wid / 3) % 8;
    int b   = wid / 24;
    int m0 = mtg * 64, n0 = nt * 16;
    const __bf16* Ap = Wc + (size_t)m0 * FEAT;
    const __bf16* Bp = Fb + ((size_t)b * NP_ + n0) * FEAT;
    v8f acc[4] = {{}, {}, {}, {}};
    for (int k0 = 0; k0 < FEAT; k0 += 32) {
        __builtin_prefetch((const void*)(Bp + k0 + 64), 0, 0);
        v16bf bm = load_ab(Bp + k0, FEAT, lane);
#pragma unroll
        for (int i = 0; i < 4; ++i) {
            v16bf a = load_ab(Ap + (size_t)(i * 16) * FEAT + k0, FEAT, lane);
            acc[i] = wmma_bf16(a, bm, acc[i]);
        }
    }
    int hi = lane >> 4;
    int n  = n0 + (lane & 15);
    if (n < N_) {
        __bf16* row = featsT + ((size_t)b * NP_ + n) * KP_ + 5;
#pragma unroll
        for (int i = 0; i < 4; ++i)
#pragma unroll
            for (int r = 0; r < 8; ++r) {
                int m = m0 + i * 16 + r + 8 * hi;
                row[m] = (__bf16)(acc[i][r] + conv_b[m]);
            }
    }
}

// ---- skew GEMMs: hx/hy[b,h,n] = sum_m W[h,m]*feats[b,m,n]  (K = KP_) --------
// wave computes 4 M-subtiles x {hx,hy}: 8 accumulators share one B fragment
__global__ void k_skew_wmma(const __bf16* __restrict__ Wx, const __bf16* __restrict__ Wy,
                            const __bf16* __restrict__ featsT,
                            float* __restrict__ hxT, float* __restrict__ hyT) {
    int wid  = blockIdx.x * 8 + (threadIdx.x >> 5);   // 128 b * 2 mtg * 3 nt waves
    int lane = threadIdx.x & 31;
    int nt  = wid % 3;
    int mtg = (wid / 3) % 2;
    int b   = wid / 6;
    int m0 = mtg * 64, n0 = nt * 16;
    const __bf16* Bp = featsT + ((size_t)b * NP_ + n0) * KP_;
    v8f ax[4] = {{}, {}, {}, {}};
    v8f ay[4] = {{}, {}, {}, {}};
    for (int k0 = 0; k0 < KP_; k0 += 32) {
        v16bf bm = load_ab(Bp + k0, KP_, lane);
#pragma unroll
        for (int i = 0; i < 4; ++i) {
            v16bf a1 = load_ab(Wx + (size_t)(m0 + i * 16) * KP_ + k0, KP_, lane);
            ax[i] = wmma_bf16(a1, bm, ax[i]);
            v16bf a2 = load_ab(Wy + (size_t)(m0 + i * 16) * KP_ + k0, KP_, lane);
            ay[i] = wmma_bf16(a2, bm, ay[i]);
        }
    }
    int hi = lane >> 4;
    int n  = n0 + (lane & 15);
    if (n < N_) {
#pragma unroll
        for (int i = 0; i < 4; ++i)
#pragma unroll
            for (int r = 0; r < 8; ++r) {
                int h = m0 + i * 16 + r + 8 * hi;
                hxT[((size_t)b * N_ + n) * HID + h] = ax[i][r];
                hyT[((size_t)b * N_ + n) * HID + h] = ay[i][r];
            }
    }
}

// ---- out[i,j] = sum_h w2[h]*relu(hx[h,i]+hy[h,j]+b1[h]);  C = out - out^T ---
__global__ void k_pairwise(const float* __restrict__ hxT, const float* __restrict__ hyT,
                           const float* __restrict__ w2, const float* __restrict__ b1,
                           float* __restrict__ Cm) {
    int b = blockIdx.x, t = threadIdx.x;      // 256 threads
    __shared__ float sw2[HID], sb1[HID], sout[N_ * N_];
    if (t < HID) { sw2[t] = w2[t]; sb1[t] = b1[t]; }
    __syncthreads();
    for (int p = t; p < N_ * N_; p += blockDim.x) {
        int i = p / N_, j = p % N_;
        const float* px = hxT + ((size_t)b * N_ + i) * HID;
        const float* py = hyT + ((size_t)b * N_ + j) * HID;
        float s = 0.0f;
        for (int h = 0; h < HID; ++h)
            s += sw2[h] * fmaxf(px[h] + py[h] + sb1[h], 0.0f);
        sout[p] = s;                           // skew_b2 cancels in C
    }
    __syncthreads();
    for (int p = t; p < N_ * N_; p += blockDim.x) {
        int i = p / N_, j = p % N_;
        Cm[(size_t)b * N_ * N_ + p] = sout[p] - sout[j * N_ + i];
    }
}

// ---- 3-step assignment iteration + final softmax ----------------------------
__global__ void k_assign(const float* __restrict__ Cm, const float* __restrict__ lr,
                         float* __restrict__ Aasg) {
    int b = blockIdx.x, t = threadIdx.x;      // 256 threads
    __shared__ float C[N_ * N_], L[N_ * N_], P[N_ * N_], M[N_ * N_];
    float lra = fabsf(lr[0]);
    for (int p = t; p < N_ * N_; p += blockDim.x) {
        C[p] = Cm[(size_t)b * N_ * N_ + p];
        L[p] = 0.0f;
    }
    __syncthreads();
    for (int step = 0; step < 3; ++step) {
        if (t < N_) {   // softmax over l for row i=t
            float mx = -1e30f;
            for (int l = 0; l < N_; ++l) mx = fmaxf(mx, L[t * N_ + l]);
            float s = 0.0f;
            for (int l = 0; l < N_; ++l) { float e = expf(L[t * N_ + l] - mx); P[t * N_ + l] = e; s += e; }
            float is = 1.0f / s;
            for (int l = 0; l < N_; ++l) P[t * N_ + l] *= is;
        }
        __syncthreads();
        if (t < N_) {   // cumsum over i for column l=t; M = above - below
            float c = 0.0f;
            for (int i = 0; i < N_; ++i) { c += P[i * N_ + t]; M[i * N_ + t] = c; }
            for (int i = 0; i < N_; ++i) {
                float cum = M[i * N_ + t];
                M[i * N_ + t] = (c - cum) - (cum - P[i * N_ + t]);
            }
        }
        __syncthreads();
        for (int p = t; p < N_ * N_; p += blockDim.x) {
            int i = p / N_, j = p % N_;
            float g = 0.0f;
            for (int l = 0; l < N_; ++l) g += M[i * N_ + l] * C[j * N_ + l];
            L[p] -= lra * g;
        }
        __syncthreads();
    }
    if (t < N_) {
        float mx = -1e30f;
        for (int l = 0; l < N_; ++l) mx = fmaxf(mx, L[t * N_ + l]);
        float s = 0.0f;
        for (int l = 0; l < N_; ++l) { float e = expf(L[t * N_ + l] - mx); P[t * N_ + l] = e; s += e; }
        float is = 1.0f / s;
        for (int l = 0; l < N_; ++l)
            Aasg[(size_t)b * N_ * N_ + t * N_ + l] = P[t * N_ + l] * is;
    }
}

// ---- x[t,b,c] = sum_l feats[b,c,l]*sig(att[b,l])*A[b,t,l]  -> bf16 [t][b][KP_]
__global__ void k_x(const __bf16* __restrict__ featsT, const float* __restrict__ satt,
                    const float* __restrict__ Aasg, __bf16* __restrict__ xbf) {
    size_t idx = (size_t)blockIdx.x * blockDim.x + threadIdx.x;
    size_t tot = (size_t)N_ * B_ * MID;
    if (idx >= tot) return;
    int c = (int)(idx % MID);
    size_t r = idx / MID;
    int b = (int)(r % B_);
    int tt = (int)(r / B_);
    float s = 0.0f;
    for (int l = 0; l < N_; ++l)
        s += (float)featsT[((size_t)b * NP_ + l) * KP_ + c] *
             satt[(size_t)b * N_ + l] *
             Aasg[((size_t)b * N_ + tt) * N_ + l];
    xbf[((size_t)tt * B_ + b) * KP_ + c] = (__bf16)s;
}

// ---- LSTM input projection: gatesX[n,m] = x[n,:]·w_ih[m,:] + b_ih + b_hh ----
// wave computes 64x16 (4 M-subtiles share each B fragment)
__global__ void k_lstm_in(const __bf16* __restrict__ Wih, const __bf16* __restrict__ xbf,
                          const float* __restrict__ bih, const float* __restrict__ bhh,
                          float* __restrict__ gatesX) {
    int wid  = blockIdx.x * 8 + (threadIdx.x >> 5);   // 64 mtg * 288 nt waves
    int lane = threadIdx.x & 31;
    int nt  = wid % 288;
    int mtg = wid / 288;
    int m0 = mtg * 64, n0 = nt * 16;
    const __bf16* Bp = xbf + (size_t)n0 * KP_;
    v8f acc[4] = {{}, {}, {}, {}};
    for (int k0 = 0; k0 < KP_; k0 += 32) {
        __builtin_prefetch((const void*)(Bp + k0 + 64), 0, 0);
        v16bf bm = load_ab(Bp + k0, KP_, lane);
#pragma unroll
        for (int i = 0; i < 4; ++i) {
            v16bf a = load_ab(Wih + (size_t)(m0 + i * 16) * KP_ + k0, KP_, lane);
            acc[i] = wmma_bf16(a, bm, acc[i]);
        }
    }
    int hi = lane >> 4;
    int n  = n0 + (lane & 15);
#pragma unroll
    for (int i = 0; i < 4; ++i)
#pragma unroll
        for (int r = 0; r < 8; ++r) {
            int m = m0 + i * 16 + r + 8 * hi;
            gatesX[(size_t)n * G4 + m] = acc[i][r] + bih[m] + bhh[m];
        }
}

// ---- fused LSTM step: GEMM + cell update, no intermediate gates buffer ------
// Wave owns u-tile u0..u0+15 (cell columns) x b-tile n0..n0+15, with its 4
// accumulators at gate-row offsets u0 + 1024*{0,1,2,3} = {i,f,g,o}.  After the
// K-loop, acc[0..3][r] are i/f/g/o for the SAME (u,b) element, so the cell
// update happens entirely in registers.  h is double-buffered (hin -> hout)
// to avoid the cross-block WAR race; c state (cst == d_out) is updated by
// exactly one lane per element.
__global__ void k_lstm_step(const __bf16* __restrict__ Whh, const __bf16* __restrict__ hin,
                            __bf16* __restrict__ hout, const float* __restrict__ gatesX,
                            float* __restrict__ cst, int t) {
    int wid  = blockIdx.x * 8 + (threadIdx.x >> 5);   // 64 utiles * 8 ntiles waves
    int lane = threadIdx.x & 31;
    int nt = wid % 8;
    int ut = wid / 8;
    int u0 = ut * 16, n0 = nt * 16;
    const __bf16* Bp = hin + (size_t)n0 * OUT_;
    v8f acc[4] = {{}, {}, {}, {}};
    for (int k0 = 0; k0 < OUT_; k0 += 32) {
        __builtin_prefetch((const void*)(Bp + k0 + 64), 0, 0);
        v16bf bm = load_ab(Bp + k0, OUT_, lane);
#pragma unroll
        for (int i = 0; i < 4; ++i) {   // gate i: rows u0 + 1024*i
            v16bf a = load_ab(Whh + (size_t)(u0 + OUT_ * i) * OUT_ + k0, OUT_, lane);
            acc[i] = wmma_bf16(a, bm, acc[i]);
        }
    }
    const float* gx = gatesX + (size_t)t * B_ * G4;
    int hi = lane >> 4;
    int b  = n0 + (lane & 15);
#pragma unroll
    for (int r = 0; r < 8; ++r) {
        int u = u0 + r + 8 * hi;
        const float* gb = gx + (size_t)b * G4 + u;
        float ig = sigf(acc[0][r] + gb[0]);
        float fg = sigf(acc[1][r] + gb[OUT_]);
        float gg = tanhf(acc[2][r] + gb[2 * OUT_]);
        float og = sigf(acc[3][r] + gb[3 * OUT_]);
        size_t cidx = (size_t)b * OUT_ + u;
        float c = fg * cst[cidx] + ig * gg;
        cst[cidx] = c;
        hout[cidx] = (__bf16)(og * tanhf(c));
    }
}

// ---------------------------------------------------------------------------
extern "C" void kernel_launch(void* const* d_in, const int* in_sizes, int n_in,
                              void* d_out, int out_size, void* d_ws, size_t ws_size,
                              hipStream_t stream) {
    const float* boxes     = (const float*)d_in[0];
    const float* attention = (const float*)d_in[1];
    const float* features  = (const float*)d_in[2];
    const float* conv_w    = (const float*)d_in[3];
    const float* conv_b    = (const float*)d_in[4];
    const float* skew_wx   = (const float*)d_in[5];
    const float* skew_wy   = (const float*)d_in[6];
    const float* skew_b1   = (const float*)d_in[7];
    const float* skew_w2   = (const float*)d_in[8];
    // d_in[9] = skew_b2 (cancels in C = out - out^T)
    const float* w_ih      = (const float*)d_in[10];
    const float* w_hh      = (const float*)d_in[11];
    const float* b_ih      = (const float*)d_in[12];
    const float* b_hh      = (const float*)d_in[13];
    const float* lr        = (const float*)d_in[14];

    char* p = (char*)d_ws;
    auto carve = [&](size_t bytes) {
        char* r = p;
        p += (bytes + 255) & ~(size_t)255;
        return (void*)r;
    };
    __bf16* featbf   = (__bf16*)carve((size_t)B_ * NP_ * FEAT * 2);
    __bf16* convw_bf = (__bf16*)carve((size_t)(MID - 5) * FEAT * 2);
    __bf16* featsT   = (__bf16*)carve((size_t)B_ * NP_ * KP_ * 2);
    __bf16* skewx_bf = (__bf16*)carve((size_t)HID * KP_ * 2);
    __bf16* skewy_bf = (__bf16*)carve((size_t)HID * KP_ * 2);
    __bf16* wih_bf   = (__bf16*)carve((size_t)G4 * KP_ * 2);
    __bf16* whh_bf   = (__bf16*)carve((size_t)G4 * OUT_ * 2);
    float*  hxT      = (float*)carve((size_t)B_ * N_ * HID * 4);
    float*  hyT      = (float*)carve((size_t)B_ * N_ * HID * 4);
    float*  Cm       = (float*)carve((size_t)B_ * N_ * N_ * 4);
    float*  Aasg     = (float*)carve((size_t)B_ * N_ * N_ * 4);
    float*  satt     = (float*)carve((size_t)B_ * N_ * 4);
    __bf16* xbf      = (__bf16*)carve((size_t)N_ * B_ * KP_ * 2);
    float*  gatesX   = (float*)carve((size_t)N_ * B_ * G4 * 4);
    __bf16* hbuf0    = (__bf16*)carve((size_t)B_ * OUT_ * 2);
    __bf16* hbuf1    = (__bf16*)carve((size_t)B_ * OUT_ * 2);
    float*  cst      = (float*)d_out;   // cell state accumulates here; final c == output
    (void)ws_size; (void)in_sizes; (void)n_in; (void)out_size;

    // zero-init padded / stateful buffers (graph-capturable memsets)
    hipMemsetAsync(featsT, 0, (size_t)B_ * NP_ * KP_ * 2, stream);
    hipMemsetAsync(xbf,    0, (size_t)N_ * B_ * KP_ * 2, stream);
    hipMemsetAsync(hbuf0,  0, (size_t)B_ * OUT_ * 2, stream);
    hipMemsetAsync(hbuf1,  0, (size_t)B_ * OUT_ * 2, stream);
    hipMemsetAsync(cst,    0, (size_t)B_ * OUT_ * 4, stream);

    // weight / activation conversion with zero padding
    auto grid1 = [](size_t n) { return (unsigned)((n + 255) / 256); };
    k_padcvt<<<grid1((size_t)(MID - 5) * FEAT), 256, 0, stream>>>(conv_w, convw_bf, MID - 5, FEAT, FEAT);
    k_padcvt<<<grid1((size_t)HID * KP_), 256, 0, stream>>>(skew_wx, skewx_bf, HID, MID, KP_);
    k_padcvt<<<grid1((size_t)HID * KP_), 256, 0, stream>>>(skew_wy, skewy_bf, HID, MID, KP_);
    k_padcvt<<<grid1((size_t)G4 * KP_), 256, 0, stream>>>(w_ih, wih_bf, G4, MID, KP_);
    k_padcvt<<<grid1((size_t)G4 * OUT_), 256, 0, stream>>>(w_hh, whh_bf, G4, OUT_, OUT_);
    k_cvt_feat<<<grid1((size_t)B_ * NP_ * FEAT), 256, 0, stream>>>(features, featbf);

    // box normalization, attention row, sigmoid(att)
    k_prep<<<B_, 64, 0, stream>>>(boxes, attention, featsT, satt);

    // conv GEMM (WMMA): 128 b * 8 mtg * 3 nt waves, 8 waves/block
    k_conv_wmma<<<(B_ * 8 * 3) / 8, 256, 0, stream>>>(convw_bf, featbf, conv_b, featsT);

    // skew GEMMs (WMMA): 128 b * 2 mtg * 3 nt waves
    k_skew_wmma<<<(B_ * 2 * 3) / 8, 256, 0, stream>>>(skewx_bf, skewy_bf, featsT, hxT, hyT);

    // pairwise relu contraction -> C, then assignment iteration
    k_pairwise<<<B_, 256, 0, stream>>>(hxT, hyT, skew_w2, skew_b1, Cm);
    k_assign<<<B_, 256, 0, stream>>>(Cm, lr, Aasg);

    // gated mixing -> LSTM input sequence (bf16, padded K)
    k_x<<<grid1((size_t)N_ * B_ * MID), 256, 0, stream>>>(featsT, satt, Aasg, xbf);

    // LSTM input projection (WMMA): 64 mtg * 288 nt waves
    k_lstm_in<<<(64 * 288) / 8, 256, 0, stream>>>(wih_bf, xbf, b_ih, b_hh, gatesX);

    // 36 fused recurrent steps (WMMA GEMM + in-register cell update),
    // h double-buffered to avoid cross-block WAR on the recurrent input
    for (int t = 0; t < N_; ++t) {
        __bf16* hin  = (t & 1) ? hbuf1 : hbuf0;
        __bf16* hout = (t & 1) ? hbuf0 : hbuf1;
        k_lstm_step<<<(64 * 8) / 8, 256, 0, stream>>>(whh_bf, hin, hout, gatesX, cst, t);
    }
}
